// NunchakuFluxTransformerBlocks_17660905521442
// MI455X (gfx1250) — compile-verified, matrix operand order checked by reading, not run on hardware
//
#include <hip/hip_runtime.h>
#include <hip/hip_bf16.h>
#include <math.h>

// ---------------------------------------------------------------------------
// Flux dual-stream transformer block for MI455X (gfx1250, wave32, WMMA).
// HBM-bound on fp32 weights (~340MB @ 23.3TB/s ~= 15us floor): stream fp32,
// convert to bf16 in-register, stage in LDS, v_wmma_f32_16x16x32_bf16.
// Attention operands are self-produced -> stored bf16 in fragment-ready
// layouts and staged with async global->LDS copies (ASYNCcnt).
// All fixed-trip loops force-unrolled so v8f element indices are static
// (avoids M0/v_movrels relative-VGPR addressing in the softmax).
// ---------------------------------------------------------------------------

typedef __attribute__((ext_vector_type(16))) __bf16 v16bf;
typedef __attribute__((ext_vector_type(8)))  float  v8f;

#define DMODEL 1536
#define S_TXT  512
#define S_IMG  1024
#define S_ALL  1536
#define NH     12
#define DH     128
#define FFDIM  6144
#define N3D    4608   // 3*D
#define N6D    9216   // 6*D
#define EPSLN  1e-6f

__device__ __forceinline__ unsigned short f2bf(float f) {
  unsigned int u = __float_as_uint(f);
  u += 0x7FFFu + ((u >> 16) & 1u);   // round-to-nearest-even-ish
  return (unsigned short)(u >> 16);
}
__device__ __forceinline__ unsigned pack2bf(float lo, float hi) {
  return (unsigned)f2bf(lo) | ((unsigned)f2bf(hi) << 16);
}

union FragAB { uint4 u4[2]; v16bf v; };  // 16 bf16 = one WMMA A/B fragment

// 32-bit LDS byte offset of a __shared__ object (generic -> as(3) cast)
__device__ __forceinline__ unsigned lds_off(const void* p) {
  return (unsigned)(uintptr_t)(__attribute__((address_space(3))) const void*)p;
}
// CDNA5 async global->LDS 16-byte copy (tracked by ASYNCcnt)
__device__ __forceinline__ void async_copy_b128(unsigned lds, const void* gp) {
  asm volatile("global_load_async_to_lds_b128 %0, %1, off"
               :: "v"(lds), "v"((unsigned long long)(uintptr_t)gp)
               : "memory");
}
__device__ __forceinline__ void wait_async0() {
  asm volatile("s_wait_asynccnt 0x0" ::: "memory");
}

// ---------------------------------------------------------------------------
// bf16-WMMA GEMM: C[M,N] = epilogue(A[M,K] @ B[K,N] + bias)
//   MODE 0: store    MODE 1: tanh-GELU    MODE 2: res + gate (.) (AB+bias)
// 64x64 block tile, 128 threads = 4 waves, each wave 32x32 (4 accumulators).
// A staged [m][k]; B staged transposed [n][k] (WMMA B-fragment layout).
// ---------------------------------------------------------------------------
template <int MODE>
__global__ __launch_bounds__(128)
void gemm_bf16(const float* __restrict__ A, const float* __restrict__ B,
               const float* __restrict__ bias, float* __restrict__ C,
               const float* __restrict__ res, const float* __restrict__ gate,
               int M, int N, int K) {
  __shared__ unsigned short As[64 * 32];
  __shared__ unsigned short Bs[64 * 32];
  const int t    = threadIdx.x;
  const int m0   = blockIdx.y * 64;
  const int n0   = blockIdx.x * 64;
  const int lane = t & 31, wave = t >> 5;
  const int wm   = (wave >> 1) * 32, wn = (wave & 1) * 32;
  const int half = lane >> 4, l15 = lane & 15;
  const int K0   = half * 8;           // per-lane K offset of fragment pattern

  v8f acc[2][2] = {};
  for (int k0 = 0; k0 < K; k0 += 32) {
    // A: float4 loads, packed bf16x2, ds_store_b64
#pragma unroll
    for (int i = 0; i < 4; i++) {
      int q = i * 128 + t;             // 512 quads
      int r = q >> 3, c = (q & 7) * 4;
      float4 a4 = *reinterpret_cast<const float4*>(&A[(size_t)(m0 + r) * K + (k0 + c)]);
      uint2 p;
      p.x = pack2bf(a4.x, a4.y);
      p.y = pack2bf(a4.z, a4.w);
      *reinterpret_cast<uint2*>(&As[r * 32 + c]) = p;
    }
    // B: coalesced-in-n loads of 2 k-rows, packed, ds_store_b32 ([n][k])
#pragma unroll
    for (int i = 0; i < 8; i++) {
      int p = i * 128 + t;             // 1024 pairs
      int n = p & 63, kk = (p >> 6) * 2;
      float b0 = B[(size_t)(k0 + kk)     * N + (n0 + n)];
      float b1 = B[(size_t)(k0 + kk + 1) * N + (n0 + n)];
      *reinterpret_cast<unsigned*>(&Bs[n * 32 + kk]) = pack2bf(b0, b1);
    }
    __syncthreads();
    FragAB af[2], bfrag[2];
#pragma unroll
    for (int mt = 0; mt < 2; mt++) {
      int base = (wm + mt * 16 + l15) * 32;
      af[mt].u4[0] = *reinterpret_cast<const uint4*>(&As[base + K0]);
      af[mt].u4[1] = *reinterpret_cast<const uint4*>(&As[base + 16 + K0]);
    }
#pragma unroll
    for (int nt = 0; nt < 2; nt++) {
      int base = (wn + nt * 16 + l15) * 32;
      bfrag[nt].u4[0] = *reinterpret_cast<const uint4*>(&Bs[base + K0]);
      bfrag[nt].u4[1] = *reinterpret_cast<const uint4*>(&Bs[base + 16 + K0]);
    }
#pragma unroll
    for (int mt = 0; mt < 2; mt++)
#pragma unroll
      for (int nt = 0; nt < 2; nt++)
        acc[mt][nt] = __builtin_amdgcn_wmma_f32_16x16x32_bf16(
            false, af[mt].v, false, bfrag[nt].v, (short)0, acc[mt][nt], false, false);
    __syncthreads();
  }
  // epilogue (C layout: element r -> row r + half*8, col = l15)
#pragma unroll
  for (int mt = 0; mt < 2; mt++)
#pragma unroll
    for (int nt = 0; nt < 2; nt++)
#pragma unroll
      for (int r = 0; r < 8; r++) {
        int row = m0 + wm + mt * 16 + r + half * 8;
        int col = n0 + wn + nt * 16 + l15;
        float v = acc[mt][nt][r] + bias[col];
        if constexpr (MODE == 1) {
          float x = v;
          v = 0.5f * x * (1.0f + tanhf(0.7978845608028654f * (x + 0.044715f * x * x * x)));
        } else if constexpr (MODE == 2) {
          v = res[(size_t)row * N + col] + gate[col] * v;
        }
        C[(size_t)row * N + col] = v;
      }
}

// ---------------------------------------------------------------------------
// LayerNorm + adaLN modulation: out = ln(x) * (1 + sc) + sh  (one row / block)
// ---------------------------------------------------------------------------
__global__ __launch_bounds__(256)
void ln_mod(const float* __restrict__ x, float* __restrict__ out,
            const float* __restrict__ emb, int sh_chunk, int sc_chunk) {
  __shared__ float red[256];
  const int row = blockIdx.x;
  const int t = threadIdx.x;
  const float* xr = x + (size_t)row * DMODEL;
  float vals[6];
  float s = 0.0f;
#pragma unroll
  for (int i = 0; i < 6; i++) { vals[i] = xr[t + i * 256]; s += vals[i]; }
  red[t] = s; __syncthreads();
  for (int o = 128; o > 0; o >>= 1) { if (t < o) red[t] += red[t + o]; __syncthreads(); }
  float mean = red[0] * (1.0f / DMODEL);
  __syncthreads();
  float vs = 0.0f;
#pragma unroll
  for (int i = 0; i < 6; i++) { float d = vals[i] - mean; vs += d * d; }
  red[t] = vs; __syncthreads();
  for (int o = 128; o > 0; o >>= 1) { if (t < o) red[t] += red[t + o]; __syncthreads(); }
  float rstd = rsqrtf(red[0] * (1.0f / DMODEL) + EPSLN);
  const float* sh = emb + sh_chunk * DMODEL;
  const float* sc = emb + sc_chunk * DMODEL;
#pragma unroll
  for (int i = 0; i < 6; i++) {
    int col = t + i * 256;
    out[(size_t)row * DMODEL + col] = (vals[i] - mean) * rstd * (1.0f + sc[col]) + sh[col];
  }
}

__global__ void silu_k(const float* __restrict__ temb, float* __restrict__ t_out) {
  int i = blockIdx.x * 256 + threadIdx.x;
  if (i < DMODEL) { float x = temb[i]; t_out[i] = x / (1.0f + expf(-x)); }
}

// adaLN GEMV: emb[0:9216]=silu(t)@Wimg+bimg, emb[9216:18432]=silu(t)@Wtxt+btxt
__global__ void adaln_k(const float* __restrict__ t, const float* __restrict__ wi,
                        const float* __restrict__ bi, const float* __restrict__ wt,
                        const float* __restrict__ bt, float* __restrict__ emb) {
  int j = blockIdx.x * 256 + threadIdx.x;
  if (j >= 2 * N6D) return;
  int which = j / N6D, col = j % N6D;
  const float* w = which ? wt : wi;
  const float* b = which ? bt : bi;
  float acc = b[col];
  for (int k = 0; k < DMODEL; k++) acc += t[k] * w[(size_t)k * N6D + col];
  emb[j] = acc;
}

// ---------------------------------------------------------------------------
// QKV post: per (head, seq) RMSNorm(q,k) + RoPE; emit bf16 operands in
// WMMA-fragment-ready layouts: q/k [h][s][dh], v transposed [h][dh][s].
// One wave per (h,s); each lane owns 4 consecutive dh (= 2 rope pairs).
// ---------------------------------------------------------------------------
__global__ __launch_bounds__(32)
void qkv_post(const float* __restrict__ qkv_txt, const float* __restrict__ qkv_img,
              const float* __restrict__ qn_t, const float* __restrict__ kn_t,
              const float* __restrict__ qn_i, const float* __restrict__ kn_i,
              const float* __restrict__ rot,
              unsigned short* __restrict__ qb, unsigned short* __restrict__ kb,
              unsigned short* __restrict__ vT) {
  const int bid = blockIdx.x;
  const int h = bid / S_ALL, s = bid % S_ALL;
  const int lane = threadIdx.x;
  const float* src; const float* qn; const float* kn;
  if (s < S_TXT) { src = qkv_txt + (size_t)s * N3D;           qn = qn_t; kn = kn_t; }
  else           { src = qkv_img + (size_t)(s - S_TXT) * N3D; qn = qn_i; kn = kn_i; }
  const int dh0 = lane * 4;
  float q[4], k[4], v[4];
#pragma unroll
  for (int j = 0; j < 4; j++) {
    q[j] = src[h * DH + dh0 + j];
    k[j] = src[DMODEL + h * DH + dh0 + j];
    v[j] = src[2 * DMODEL + h * DH + dh0 + j];
  }
  float sq = 0.0f, sk = 0.0f;
#pragma unroll
  for (int j = 0; j < 4; j++) { sq += q[j] * q[j]; sk += k[j] * k[j]; }
#pragma unroll
  for (int m = 1; m < 32; m <<= 1) { sq += __shfl_xor(sq, m, 32); sk += __shfl_xor(sk, m, 32); }
  float rq = rsqrtf(sq * (1.0f / DH) + EPSLN);
  float rk = rsqrtf(sk * (1.0f / DH) + EPSLN);
#pragma unroll
  for (int j = 0; j < 4; j++) { q[j] *= rq * qn[dh0 + j]; k[j] *= rk * kn[dh0 + j]; }
  const size_t ob = ((size_t)h * S_ALL + s) * DH + dh0;
#pragma unroll
  for (int p = 0; p < 2; p++) {
    int i = lane * 2 + p;
    const float* r = rot + ((size_t)s * 64 + i) * 4;   // [s][64][2][2]
    float q0 = q[2 * p], q1 = q[2 * p + 1];
    float k0 = k[2 * p], k1 = k[2 * p + 1];
    qb[ob + 2 * p]     = f2bf(r[0] * q0 + r[1] * q1);
    qb[ob + 2 * p + 1] = f2bf(r[2] * q0 + r[3] * q1);
    kb[ob + 2 * p]     = f2bf(r[0] * k0 + r[1] * k1);
    kb[ob + 2 * p + 1] = f2bf(r[2] * k0 + r[3] * k1);
    vT[((size_t)h * DH + dh0 + 2 * p)     * S_ALL + s] = f2bf(v[2 * p]);
    vT[((size_t)h * DH + dh0 + 2 * p + 1) * S_ALL + s] = f2bf(v[2 * p + 1]);
  }
}

// ---------------------------------------------------------------------------
// Flash attention: block = (head, 64-query tile), 4 waves x 16 q-rows.
// Q/K/V tiles are raw bf16 copies -> async global->LDS (ASYNCcnt) staging.
// Scores + P@V via WMMA; online softmax in f32; P reshaped through LDS.
// ---------------------------------------------------------------------------
__global__ __launch_bounds__(128)
void attn_k(const unsigned short* __restrict__ qg, const unsigned short* __restrict__ kg,
            const unsigned short* __restrict__ vg, float* __restrict__ og) {
  __shared__ unsigned short Qs[64 * DH];      // [m][d]
  __shared__ unsigned short Ks[32 * DH];      // [key][d]   (B-frag for Q@K^T)
  __shared__ unsigned short Vs[DH * 32];      // [d][key]   (B-frag for P@V)
  __shared__ unsigned short Ps[4][16 * 32];   // per-wave P [m][key]
  const int t = threadIdx.x;
  const int h = blockIdx.x / 24;
  const int q0 = (blockIdx.x % 24) * 64;
  const int lane = t & 31, wave = t >> 5;
  const int half = lane >> 4, l15 = lane & 15;
  const int K0 = half * 8;
  const unsigned short* qh = qg + (size_t)h * S_ALL * DH;
  const unsigned short* kh = kg + (size_t)h * S_ALL * DH;
  const unsigned short* vh = vg + (size_t)h * DH * S_ALL;  // transposed [dh][s]

  // Q tile: 16KB contiguous copy, 128B per thread (8 async b128)
  {
    unsigned qdst = lds_off(Qs) + t * 128;
    const char* qsrc = (const char*)(qh + (size_t)q0 * DH) + t * 128;
#pragma unroll
    for (int i = 0; i < 8; i++) async_copy_b128(qdst + i * 16, qsrc + i * 16);
  }
  v8f o[8] = {};
  float mr[8], lr[8];
#pragma unroll
  for (int r = 0; r < 8; r++) { mr[r] = -1e30f; lr[r] = 0.0f; }
  const float scale = 0.08838834764831845f;   // 1/sqrt(128)

  for (int kt = 0; kt < S_ALL; kt += 32) {
    __syncthreads();                          // previous iter done with Ks/Vs
    {
      // K tile: 8KB contiguous, 64B per thread
      unsigned kdst = lds_off(Ks) + t * 64;
      const char* ksrc = (const char*)(kh + (size_t)kt * DH) + t * 64;
#pragma unroll
      for (int i = 0; i < 4; i++) async_copy_b128(kdst + i * 16, ksrc + i * 16);
      // V tile: thread t copies row d=t of V^T (64B)
      unsigned vdst = lds_off(Vs) + t * 64;
      const char* vsrc = (const char*)(vh + (size_t)t * S_ALL + kt);
#pragma unroll
      for (int i = 0; i < 4; i++) async_copy_b128(vdst + i * 16, vsrc + i * 16);
    }
    wait_async0();                            // covers Q on first iteration too
    __syncthreads();
    // two 16x16 score tiles = Q(16x128) @ K^T via 4 WMMAs each
    v8f sc[2];
#pragma unroll
    for (int kk = 0; kk < 2; kk++) {
      v8f a = {};
#pragma unroll
      for (int kd = 0; kd < DH; kd += 32) {
        FragAB qf, kf;
        int qb = (wave * 16 + l15) * DH + kd;
        qf.u4[0] = *reinterpret_cast<const uint4*>(&Qs[qb + K0]);
        qf.u4[1] = *reinterpret_cast<const uint4*>(&Qs[qb + 16 + K0]);
        int kb = (kk * 16 + l15) * DH + kd;
        kf.u4[0] = *reinterpret_cast<const uint4*>(&Ks[kb + K0]);
        kf.u4[1] = *reinterpret_cast<const uint4*>(&Ks[kb + 16 + K0]);
        a = __builtin_amdgcn_wmma_f32_16x16x32_bf16(false, qf.v, false, kf.v,
                                                    (short)0, a, false, false);
      }
      sc[kk] = a;
    }
    // online softmax: row m = r + half*8 lives in element r across 16 lanes
#pragma unroll
    for (int r = 0; r < 8; r++) {
      float s0 = sc[0][r] * scale, s1 = sc[1][r] * scale;
      float tm = fmaxf(s0, s1);
#pragma unroll
      for (int m = 1; m < 16; m <<= 1) tm = fmaxf(tm, __shfl_xor(tm, m, 32));
      float mnew = fmaxf(mr[r], tm);
      float fac = expf(mr[r] - mnew);
      float p0 = expf(s0 - mnew), p1 = expf(s1 - mnew);
      float ps = p0 + p1;
#pragma unroll
      for (int m = 1; m < 16; m <<= 1) ps += __shfl_xor(ps, m, 32);
      lr[r] = lr[r] * fac + ps;
      mr[r] = mnew;
#pragma unroll
      for (int j = 0; j < 8; j++) o[j][r] *= fac;
      int mrow = r + half * 8;
      Ps[wave][mrow * 32 + l15]      = f2bf(p0);
      Ps[wave][mrow * 32 + 16 + l15] = f2bf(p1);
    }
    // per-wave LDS RAW: drain DScnt before re-reading own P region
    asm volatile("s_wait_dscnt 0" ::: "memory");
    FragAB pf;
    int pb = l15 * 32;
    pf.u4[0] = *reinterpret_cast<const uint4*>(&Ps[wave][pb + K0]);
    pf.u4[1] = *reinterpret_cast<const uint4*>(&Ps[wave][pb + 16 + K0]);
#pragma unroll
    for (int j = 0; j < 8; j++) {             // O(16x128) += P(16x32) @ V(32x128)
      FragAB vf;
      int vb = (j * 16 + l15) * 32;
      vf.u4[0] = *reinterpret_cast<const uint4*>(&Vs[vb + K0]);
      vf.u4[1] = *reinterpret_cast<const uint4*>(&Vs[vb + 16 + K0]);
      o[j] = __builtin_amdgcn_wmma_f32_16x16x32_bf16(false, pf.v, false, vf.v,
                                                     (short)0, o[j], false, false);
    }
  }
#pragma unroll
  for (int r = 0; r < 8; r++) {
    float inv = 1.0f / lr[r];
    int row = q0 + wave * 16 + r + half * 8;
#pragma unroll
    for (int j = 0; j < 8; j++)
      og[(size_t)row * DMODEL + h * DH + j * 16 + l15] = o[j][r] * inv;
  }
}

// ---------------------------------------------------------------------------
extern "C" void kernel_launch(void* const* d_in, const int* in_sizes, int n_in,
                              void* d_out, int out_size, void* d_ws, size_t ws_size,
                              hipStream_t stream) {
  const float* hidden = (const float*)d_in[0];
  const float* enc    = (const float*)d_in[1];
  const float* temb   = (const float*)d_in[2];
  const float* rot    = (const float*)d_in[3];
  const float* aiw = (const float*)d_in[4];
  const float* aib = (const float*)d_in[5];
  const float* atw = (const float*)d_in[6];
  const float* atb = (const float*)d_in[7];
  const float* qiw = (const float*)d_in[8];
  const float* qib = (const float*)d_in[9];
  const float* qtw = (const float*)d_in[10];
  const float* qtb = (const float*)d_in[11];
  const float* qni = (const float*)d_in[12];
  const float* kni = (const float*)d_in[13];
  const float* qnt = (const float*)d_in[14];
  const float* knt = (const float*)d_in[15];
  const float* oiw = (const float*)d_in[16];
  const float* oib = (const float*)d_in[17];
  const float* otw = (const float*)d_in[18];
  const float* otb = (const float*)d_in[19];
  const float* mi1w = (const float*)d_in[20];
  const float* mi1b = (const float*)d_in[21];
  const float* mi2w = (const float*)d_in[22];
  const float* mi2b = (const float*)d_in[23];
  const float* mt1w = (const float*)d_in[24];
  const float* mt1b = (const float*)d_in[25];
  const float* mt2w = (const float*)d_in[26];
  const float* mt2b = (const float*)d_in[27];

  float* ws = (float*)d_ws;
  size_t off = 0;
  float* t_silu = ws + off; off += DMODEL;
  float* emb    = ws + off; off += (size_t)2 * N6D;        // [img | txt]
  float* n_txt  = ws + off; off += (size_t)S_TXT * DMODEL;
  float* n_img  = ws + off; off += (size_t)S_IMG * DMODEL;
  float* qkv_t  = ws + off; off += (size_t)S_TXT * N3D;
  float* qkv_i  = ws + off; off += (size_t)S_IMG * N3D;
  unsigned short* qbuf = (unsigned short*)(ws + off); off += (size_t)NH * S_ALL * DH / 2;
  unsigned short* kbuf = (unsigned short*)(ws + off); off += (size_t)NH * S_ALL * DH / 2;
  unsigned short* vTbf = (unsigned short*)(ws + off); off += (size_t)NH * S_ALL * DH / 2;
  float* obuf   = ws + off; off += (size_t)S_ALL * DMODEL;
  float* h2_t   = ws + off; off += (size_t)S_TXT * DMODEL;
  float* h2_i   = ws + off; off += (size_t)S_IMG * DMODEL;
  float* n2_t   = ws + off; off += (size_t)S_TXT * DMODEL;
  float* n2_i   = ws + off; off += (size_t)S_IMG * DMODEL;
  float* ff_t   = ws + off; off += (size_t)S_TXT * FFDIM;
  float* ff_i   = ws + off; off += (size_t)S_IMG * FFDIM;

  float* out_txt = (float*)d_out;
  float* out_img = (float*)d_out + (size_t)S_TXT * DMODEL;
  float* emb_i = emb;
  float* emb_t = emb + N6D;

  // temb path
  silu_k<<<(DMODEL + 255) / 256, 256, 0, stream>>>(temb, t_silu);
  adaln_k<<<(2 * N6D + 255) / 256, 256, 0, stream>>>(t_silu, aiw, aib, atw, atb, emb);

  // LN1 + modulation (chunks: 0=shift, 1=scale)
  ln_mod<<<S_TXT, 256, 0, stream>>>(enc,    n_txt, emb_t, 0, 1);
  ln_mod<<<S_IMG, 256, 0, stream>>>(hidden, n_img, emb_i, 0, 1);

  // QKV projections (WMMA)
  gemm_bf16<0><<<dim3(N3D / 64, S_TXT / 64), 128, 0, stream>>>(n_txt, qtw, qtb, qkv_t, nullptr, nullptr, S_TXT, N3D, DMODEL);
  gemm_bf16<0><<<dim3(N3D / 64, S_IMG / 64), 128, 0, stream>>>(n_img, qiw, qib, qkv_i, nullptr, nullptr, S_IMG, N3D, DMODEL);

  // RMSNorm + RoPE + head split -> bf16 fragment-ready buffers
  qkv_post<<<NH * S_ALL, 32, 0, stream>>>(qkv_t, qkv_i, qnt, knt, qni, kni, rot, qbuf, kbuf, vTbf);

  // Flash attention (WMMA + async global->LDS staging)
  attn_k<<<NH * (S_ALL / 64), 128, 0, stream>>>(qbuf, kbuf, vTbf, obuf);

  // Output projections fused with residual + gate (chunk 2 = g)
  gemm_bf16<2><<<dim3(DMODEL / 64, S_TXT / 64), 128, 0, stream>>>(obuf, otw, otb, h2_t, enc, emb_t + 2 * DMODEL, S_TXT, DMODEL, DMODEL);
  gemm_bf16<2><<<dim3(DMODEL / 64, S_IMG / 64), 128, 0, stream>>>(obuf + (size_t)S_TXT * DMODEL, oiw, oib, h2_i, hidden, emb_i + 2 * DMODEL, S_IMG, DMODEL, DMODEL);

  // LN2 + modulation (chunks: 3=shift_mlp, 4=scale_mlp)
  ln_mod<<<S_TXT, 256, 0, stream>>>(h2_t, n2_t, emb_t, 3, 4);
  ln_mod<<<S_IMG, 256, 0, stream>>>(h2_i, n2_i, emb_i, 3, 4);

  // MLP up + GELU (fused epilogue)
  gemm_bf16<1><<<dim3(FFDIM / 64, S_TXT / 64), 128, 0, stream>>>(n2_t, mt1w, mt1b, ff_t, nullptr, nullptr, S_TXT, FFDIM, DMODEL);
  gemm_bf16<1><<<dim3(FFDIM / 64, S_IMG / 64), 128, 0, stream>>>(n2_i, mi1w, mi1b, ff_i, nullptr, nullptr, S_IMG, FFDIM, DMODEL);

  // MLP down fused with residual + gate (chunk 5 = gate_mlp) -> final out
  gemm_bf16<2><<<dim3(DMODEL / 64, S_TXT / 64), 128, 0, stream>>>(ff_t, mt2w, mt2b, out_txt, h2_t, emb_t + 5 * DMODEL, S_TXT, DMODEL, FFDIM);
  gemm_bf16<2><<<dim3(DMODEL / 64, S_IMG / 64), 128, 0, stream>>>(ff_i, mi2w, mi2b, out_img, h2_i, emb_i + 5 * DMODEL, S_IMG, DMODEL, FFDIM);

  (void)in_sizes; (void)n_in; (void)out_size; (void)ws_size;
}
